// TrackOn_28166395527484
// MI455X (gfx1250) — matrix operand order repaired and checked by reading, not compiled, not run on hardware
//
#include <hip/hip_runtime.h>

typedef __attribute__((ext_vector_type(16))) _Float16 v16h;
typedef __attribute__((ext_vector_type(8)))  _Float16 v8h;
typedef __attribute__((ext_vector_type(8)))  float    v8f;

#define C_DIM 256
#define HP 96
#define WP 128
#define P0 (HP*WP)      // 12288
#define NQ 256
#define BATCH 4

// -------------------------------------------------------------------------
// Kernel 1: k x k average pool + L2-normalize over channels + f32 -> f16.
// Output layout: [batch][pos][C] (position-major; contiguous channel writes,
// and exactly the layout the WMMA B-fragment wants to load contiguously).
// One wave (32 lanes) per output position; each lane owns 8 channels.
// -------------------------------------------------------------------------
__global__ void pool_norm_f16(const float* __restrict__ src, _Float16* __restrict__ dst,
                              int nposPerBatch, int wSrc, int k,
                              long srcBatchStride, long dstBatchStride) {
    int wid  = (blockIdx.x * blockDim.x + threadIdx.x) >> 5;
    int lane = threadIdx.x & 31;
    int b = wid / nposPerBatch;
    int p = wid % nposPerBatch;
    int wOut = wSrc / k;
    int y = p / wOut, x = p % wOut;
    const float* sb = src + (long)b * srcBatchStride;
    int c0 = lane * 8;

    float acc[8];
#pragma unroll
    for (int i = 0; i < 8; ++i) acc[i] = 0.f;

    for (int dy = 0; dy < k; ++dy) {
        for (int dx = 0; dx < k; ++dx) {
            long sp = (long)(y * k + dy) * wSrc + (x * k + dx);
            const float* rp = sb + sp * C_DIM + c0;
            float4 a = *(const float4*)rp;
            float4 c = *(const float4*)(rp + 4);
            acc[0] += a.x; acc[1] += a.y; acc[2] += a.z; acc[3] += a.w;
            acc[4] += c.x; acc[5] += c.y; acc[6] += c.z; acc[7] += c.w;
        }
    }
    float scale = 1.0f / (float)(k * k);
    float ss = 0.f;
#pragma unroll
    for (int i = 0; i < 8; ++i) { acc[i] *= scale; ss += acc[i] * acc[i]; }
    // wave32 reduction
    for (int m = 16; m > 0; m >>= 1) ss += __shfl_xor(ss, m, 32);
    float inv = 1.0f / fmaxf(sqrtf(ss), 1e-12f);

    v8h outv;
#pragma unroll
    for (int i = 0; i < 8; ++i) outv[i] = (_Float16)(acc[i] * inv);
    *(v8h*)(dst + (long)b * dstBatchStride + (long)p * C_DIM + c0) = outv;
}

// -------------------------------------------------------------------------
// Kernel 2: correlation GEMM via WMMA f16->f32.
//   A = qn   [B][256][256]  (row-major, M x K)
//   B = fln  [B][P][256]    (position-major => contiguous K per column)
//   C = corr [B][256][P]    (f32)
// One wave computes a 32(M) x 64(N) tile: 2 A fragments x 4 B fragments,
// 8 accumulators. K=256 swept in 8 steps of 32 -> 64 WMMAs per wave at
// 1.5 b128-loads per WMMA (B reuse doubled vs a 16-row strip).
// -------------------------------------------------------------------------
__global__ void corr_wmma(const _Float16* __restrict__ Aq,
                          const _Float16* __restrict__ Bf,
                          float* __restrict__ Cc,
                          int P) {
    int wid  = (blockIdx.x * blockDim.x + threadIdx.x) >> 5;
    int lane = threadIdx.x & 31;
    int tilesN = P >> 6;                 // 64-wide N strips
    int b = wid / (8 * tilesN);          // 8 M-tiles of 32 rows
    int r = wid % (8 * tilesN);
    int mBase   = (r / tilesN) * 32;
    int posBase = (r % tilesN) * 64;
    int hi16 = (lane < 16) ? 0 : 1;
    int col  = lane & 15;

    // A fragment addressing (16-bit A 16x32, this code's own k order, not the ISA table): row = base+lane%16,
    // elements 0..15 at K = kb + 16*hi16 .. +15 contiguous (matches B).
    const _Float16* Ab0 = Aq + ((long)b * NQ + (mBase      + col)) * C_DIM + hi16 * 16;
    const _Float16* Ab1 = Aq + ((long)b * NQ + (mBase + 16 + col)) * C_DIM + hi16 * 16;
    // B fragment: lane holds column (posBase + t*16 + col), K = kb+16*hi16..+15 contiguous.
    const _Float16* Bb = Bf + (long)b * P * C_DIM + hi16 * 16;

    v8f acc00 = {}, acc01 = {}, acc02 = {}, acc03 = {};
    v8f acc10 = {}, acc11 = {}, acc12 = {}, acc13 = {};

#pragma unroll
    for (int kb = 0; kb < C_DIM; kb += 32) {
        v8h lo0 = *(const v8h*)(Ab0 + kb);
        v8h hh0 = *(const v8h*)(Ab0 + kb + 8);
        v8h lo1 = *(const v8h*)(Ab1 + kb);
        v8h hh1 = *(const v8h*)(Ab1 + kb + 8);
        v16h a0, a1;
#pragma unroll
        for (int i = 0; i < 8; ++i) {
            a0[i] = lo0[i]; a0[i + 8] = hh0[i];
            a1[i] = lo1[i]; a1[i + 8] = hh1[i];
        }

        const _Float16* bp0 = Bb + (long)(posBase      + col) * C_DIM + kb;
        const _Float16* bp1 = Bb + (long)(posBase + 16 + col) * C_DIM + kb;
        const _Float16* bp2 = Bb + (long)(posBase + 32 + col) * C_DIM + kb;
        const _Float16* bp3 = Bb + (long)(posBase + 48 + col) * C_DIM + kb;
        __builtin_prefetch(bp0 + 32, 0, 1);   // global_prefetch_b8 next K block
        v16h b0 = *(const v16h*)bp0;
        v16h b1 = *(const v16h*)bp1;
        v16h b2 = *(const v16h*)bp2;
        v16h b3 = *(const v16h*)bp3;

        acc00 = __builtin_amdgcn_wmma_f32_16x16x32_f16(false, a0, false, b0, (short)0, acc00, false, false);
        acc10 = __builtin_amdgcn_wmma_f32_16x16x32_f16(false, a1, false, b0, (short)0, acc10, false, false);
        acc01 = __builtin_amdgcn_wmma_f32_16x16x32_f16(false, a0, false, b1, (short)0, acc01, false, false);
        acc11 = __builtin_amdgcn_wmma_f32_16x16x32_f16(false, a1, false, b1, (short)0, acc11, false, false);
        acc02 = __builtin_amdgcn_wmma_f32_16x16x32_f16(false, a0, false, b2, (short)0, acc02, false, false);
        acc12 = __builtin_amdgcn_wmma_f32_16x16x32_f16(false, a1, false, b2, (short)0, acc12, false, false);
        acc03 = __builtin_amdgcn_wmma_f32_16x16x32_f16(false, a0, false, b3, (short)0, acc03, false, false);
        acc13 = __builtin_amdgcn_wmma_f32_16x16x32_f16(false, a1, false, b3, (short)0, acc13, false, false);
    }

    // D layout: VGPR r holds M = base + r + 8*hi16, N = posBase + col.
    float* Cb0 = Cc + ((long)b * NQ + mBase      + hi16 * 8) * P + posBase + col;
    float* Cb1 = Cc + ((long)b * NQ + mBase + 16 + hi16 * 8) * P + posBase + col;
#pragma unroll
    for (int r8 = 0; r8 < 8; ++r8) {
        Cb0[(long)r8 * P +  0] = acc00[r8];
        Cb0[(long)r8 * P + 16] = acc01[r8];
        Cb0[(long)r8 * P + 32] = acc02[r8];
        Cb0[(long)r8 * P + 48] = acc03[r8];
        Cb1[(long)r8 * P +  0] = acc10[r8];
        Cb1[(long)r8 * P + 16] = acc11[r8];
        Cb1[(long)r8 * P + 32] = acc12[r8];
        Cb1[(long)r8 * P + 48] = acc13[r8];
    }
}

// -------------------------------------------------------------------------
// Kernel 3: fused bilinear-upsample + conv3x3(4->16) + relu + conv3x3(16->1)
// + residual (corr level 0). One 16x16 output tile per 256-thread block.
// Stages the 4-channel c-tile (halo 2) and the 16-channel hmid tile (halo 1)
// in LDS so neither ever hits HBM.
// -------------------------------------------------------------------------
__global__ void fused_conv(const float* __restrict__ corr0,
                           const float* __restrict__ corr1,
                           const float* __restrict__ corr2,
                           const float* __restrict__ corr3,
                           const float* __restrict__ w1,
                           const float* __restrict__ w2,
                           float* __restrict__ out) {
    __shared__ float sc[4][20][20];     // c tile, halo 2
    __shared__ float sh[16][18][18];    // hmid tile, halo 1
    __shared__ float sw1[576];
    __shared__ float sw2[144];

    int bi   = blockIdx.x;
    int tile = bi % 48;                 // 6 x 8 tiles of 16x16
    int img  = bi / 48;                 // b*256 + n
    int tileY = (tile / 8) * 16;
    int tileX = (tile % 8) * 16;
    int tid = threadIdx.x;

    if (tid < 576) sw1[tid] = w1[tid];
    if (tid < 144) sw2[tid] = w2[tid];

    const float* cl0 = corr0 + (long)img * P0;
    const float* cl1 = corr1 + (long)img * (P0 / 4);
    const float* cl2 = corr2 + (long)img * (P0 / 16);
    const float* cl3 = corr3 + (long)img * (P0 / 64);

    // ---- stage c (4 channels, 20x20 with zero pad + on-the-fly bilinear) ----
    for (int i = tid; i < 1600; i += 256) {
        int ch = i / 400;
        int j  = i % 400;
        int sy = j / 20, sx = j % 20;
        int yy = tileY - 2 + sy;
        int xx = tileX - 2 + sx;
        float v = 0.f;
        if (yy >= 0 && yy < HP && xx >= 0 && xx < WP) {
            if (ch == 0) {
                v = cl0[yy * WP + xx];
            } else {
                int k  = 1 << ch;
                int hl = HP >> ch, wl = WP >> ch;
                const float* cp = (ch == 1) ? cl1 : (ch == 2) ? cl2 : cl3;
                // half-pixel linear resize (torch align_corners=False)
                float sxf = fminf(fmaxf((xx + 0.5f) / (float)k - 0.5f, 0.f), (float)(wl - 1));
                float syf = fminf(fmaxf((yy + 0.5f) / (float)k - 0.5f, 0.f), (float)(hl - 1));
                int x0 = (int)sxf, y0 = (int)syf;
                float fx = sxf - (float)x0, fy = syf - (float)y0;
                int x1 = min(x0 + 1, wl - 1), y1 = min(y0 + 1, hl - 1);
                float v00 = cp[y0 * wl + x0], v01 = cp[y0 * wl + x1];
                float v10 = cp[y1 * wl + x0], v11 = cp[y1 * wl + x1];
                v = (v00 * (1.f - fx) + v01 * fx) * (1.f - fy)
                  + (v10 * (1.f - fx) + v11 * fx) * fy;
            }
        }
        sc[ch][sy][sx] = v;
    }
    __syncthreads();

    // ---- conv1 + relu -> hmid (zero outside the image, matching SAME pad) ----
    for (int i = tid; i < 5184; i += 256) {
        int oc = i / 324;
        int j  = i % 324;
        int hy = j / 18, hx = j % 18;
        int fy = tileY - 1 + hy, fx = tileX - 1 + hx;
        float s = 0.f;
        if (fy >= 0 && fy < HP && fx >= 0 && fx < WP) {
#pragma unroll
            for (int ic = 0; ic < 4; ++ic)
#pragma unroll
                for (int ky = 0; ky < 3; ++ky)
#pragma unroll
                    for (int kx = 0; kx < 3; ++kx)
                        s += sw1[((oc * 4 + ic) * 3 + ky) * 3 + kx] * sc[ic][hy + ky][hx + kx];
            s = fmaxf(s, 0.f);
        }
        sh[oc][hy][hx] = s;
    }
    __syncthreads();

    // ---- conv2 + residual ----
    int oy = tid / 16, ox = tid % 16;
    float s = 0.f;
#pragma unroll
    for (int ic = 0; ic < 16; ++ic)
#pragma unroll
        for (int ky = 0; ky < 3; ++ky)
#pragma unroll
            for (int kx = 0; kx < 3; ++kx)
                s += sw2[(ic * 3 + ky) * 3 + kx] * sh[ic][oy + ky][ox + kx];
    s += sc[0][oy + 2][ox + 2];
    int y = tileY + oy, x = tileX + ox;
    out[(long)img * P0 + y * WP + x] = s;
}

// -------------------------------------------------------------------------
extern "C" void kernel_launch(void* const* d_in, const int* in_sizes, int n_in,
                              void* d_out, int out_size, void* d_ws, size_t ws_size,
                              hipStream_t stream) {
    (void)in_sizes; (void)n_in; (void)out_size; (void)ws_size;
    const float* q_t = (const float*)d_in[0];   // [4,256,256]
    const float* f_t = (const float*)d_in[1];   // [4,12288,256]
    const float* w1  = (const float*)d_in[2];   // [16,4,3,3]
    const float* w2  = (const float*)d_in[3];   // [1,16,3,3]
    float* out = (float*)d_out;                 // [4,256,12288]

    char* ws = (char*)d_ws;
    _Float16* qn   = (_Float16*)(ws);                 // 512 KB
    _Float16* fln0 = (_Float16*)(ws + 524288UL);      // 24 MB
    _Float16* fln1 = (_Float16*)(ws + 25690112UL);    // 6 MB
    _Float16* fln2 = (_Float16*)(ws + 31981568UL);    // 1.5 MB
    _Float16* fln3 = (_Float16*)(ws + 33554432UL);    // 384 KB
    float* corr0 = (float*)(ws + 33947648UL);         // 48 MB
    float* corr1 = (float*)(ws + 84279296UL);         // 12 MB
    float* corr2 = (float*)(ws + 96862208UL);         // 3 MB
    float* corr3 = (float*)(ws + 100007936UL);        // 768 KB

    // ---- normalize q (k=1 pooling over a 1x256 "image") ----
    pool_norm_f16<<<(BATCH * NQ) / 8, 256, 0, stream>>>(
        q_t, qn, NQ, NQ, 1, (long)NQ * C_DIM, (long)NQ * C_DIM);

    // ---- pyramid levels: pool + channel-normalize -> f16 ----
    const int levP[4] = { P0, P0 / 4, P0 / 16, P0 / 64 };
    _Float16* levF[4] = { fln0, fln1, fln2, fln3 };
    for (int l = 0; l < 4; ++l) {
        int k = 1 << l;
        int np = levP[l];
        pool_norm_f16<<<(BATCH * np) / 8, 256, 0, stream>>>(
            f_t, levF[l], np, WP, k, (long)P0 * C_DIM, (long)np * C_DIM);
    }

    // ---- correlation GEMMs (WMMA), 32x64 tile per wave ----
    float* levC[4] = { corr0, corr1, corr2, corr3 };
    for (int l = 0; l < 4; ++l) {
        int P = levP[l];
        int waves = BATCH * 8 * (P / 64);     // 8 M-tiles (32 rows) x (P/64) strips
        corr_wmma<<<waves / 8, 256, 0, stream>>>(qn, levF[l], levC[l], P);
    }

    // ---- fused upsample + conv + residual ----
    fused_conv<<<BATCH * NQ * 48, 256, 0, stream>>>(
        corr0, corr1, corr2, corr3, w1, w2, out);
}
